// CWRNNLM_18597208391928
// MI455X (gfx1250) — matrix operand order repaired and checked by reading, not compiled, not run on hardware
//
#include <hip/hip_runtime.h>
#include <hip/hip_bf16.h>
#include <math.h>

// ---------------- CDNA5 WMMA types ----------------
typedef __bf16 bf16_t;
typedef __attribute__((ext_vector_type(16))) __bf16 v16bf;
typedef __attribute__((ext_vector_type(8)))  float  v8f;
typedef __attribute__((ext_vector_type(4)))  int    v4i;

#define NH      1024
#define EMB     512
#define VOCAB   32000
#define BATCH   16
#define TSTEPS  256
#define TFULL   257

// -------- optional CDNA5 async global->LDS copy (ASYNCcnt path) ----------
#if defined(__has_builtin)
#if __has_builtin(__builtin_amdgcn_global_load_async_to_lds_b128) && \
    __has_builtin(__builtin_amdgcn_s_wait_asynccnt)
#define HAVE_ASYNC_COPY 1
#endif
#endif

#ifdef HAVE_ASYNC_COPY
typedef __attribute__((address_space(1))) v4i* gptr_v4i;   // global
typedef __attribute__((address_space(3))) v4i* lptr_v4i;   // LDS
#endif

__device__ __forceinline__ void copy16B_to_lds(void* lds_dst, const void* gsrc) {
#ifdef HAVE_ASYNC_COPY
    __builtin_amdgcn_global_load_async_to_lds_b128(
        (gptr_v4i)gsrc, (lptr_v4i)lds_dst, 0, 0);
#else
    *(uint4*)lds_dst = *(const uint4*)gsrc;
#endif
}
__device__ __forceinline__ void async_copy_wait() {
#ifdef HAVE_ASYNC_COPY
    __builtin_amdgcn_s_wait_asynccnt(0);
#endif
}

__device__ __forceinline__ bf16_t f2bf(float f) {
    union { float f; unsigned u; } v; v.f = f;
    unsigned r = v.u + 0x7FFFu + ((v.u >> 16) & 1u);   // round-to-nearest-even
    unsigned short h = (unsigned short)(r >> 16);
    bf16_t out; __builtin_memcpy(&out, &h, 2); return out;
}

// convert 8 contiguous fp32 -> 8 bf16, one 16B store
__device__ __forceinline__ void cvt8_store(const float* __restrict__ s,
                                           bf16_t* __restrict__ d) {
    float4 a = *(const float4*)s;
    float4 b = *(const float4*)(s + 4);
    union { bf16_t h[8]; uint4 q; } u;
    u.h[0] = f2bf(a.x); u.h[1] = f2bf(a.y); u.h[2] = f2bf(a.z); u.h[3] = f2bf(a.w);
    u.h[4] = f2bf(b.x); u.h[5] = f2bf(b.y); u.h[6] = f2bf(b.z); u.h[7] = f2bf(b.w);
    *(uint4*)d = u.q;
}

// ---------------- embedding gather -> bf16 X (rows ordered m = t*B + b) ----
__global__ void k_embed_gather(const int* __restrict__ x,
                               const float* __restrict__ emb,
                               bf16_t* __restrict__ Xbf) {
    const long n8 = (long)BATCH * TSTEPS * (EMB / 8);
    for (long i = (long)blockIdx.x * blockDim.x + threadIdx.x; i < n8;
         i += (long)gridDim.x * blockDim.x) {
        long m = i / (EMB / 8); int k = (int)(i - m * (EMB / 8)) * 8;
        int b = (int)(m % BATCH), t = (int)(m / BATCH);
        int tok = x[b * TFULL + t];               // xin = x[:, :-1]
        cvt8_store(&emb[(long)tok * EMB + k], &Xbf[m * EMB + k]);
    }
}

// ---------------- fp32 -> bf16 convert, 8-wide, no mask --------------------
__global__ void k_cvt8(const float* __restrict__ src, bf16_t* __restrict__ dst,
                       long n8) {
    for (long i = (long)blockIdx.x * blockDim.x + threadIdx.x; i < n8;
         i += (long)gridDim.x * blockDim.x)
        cvt8_store(src + i * 8, dst + i * 8);
}

// ------- Wh: block-tri mask + TRANSPOSE into fragment-linear layout --------
// WhT[c * NH + k] = mask(Wh[k * NH + c]) ; scan B-fragments become contiguous.
__global__ void k_cvt_whT(const float* __restrict__ Wh, bf16_t* __restrict__ WhT) {
    const long n = (long)NH * NH;
    for (long i = (long)blockIdx.x * blockDim.x + threadIdx.x; i < n;
         i += (long)gridDim.x * blockDim.x) {
        int r = (int)(i >> 10), c = (int)(i & (NH - 1));
        float v = Wh[i];
        if ((c >> 8) < (r >> 8)) v = 0.f;         // keep block[c] >= block[r]
        WhT[(long)c * NH + r] = f2bf(v);
    }
}

// ---------------- generic bf16 WMMA GEMM: C(MxN) = A(MxK) @ B(KxN) (+bias) --
// 256 threads = 8 waves; block tile 128x128; wave subtile 32x64.
// A staged row-major (async global->LDS), B staged transposed for b128 frags.
__global__ __launch_bounds__(256)
void k_gemm_bf16(const bf16_t* __restrict__ A, const bf16_t* __restrict__ B,
                 float* __restrict__ C, int M, int N, int K,
                 const float* __restrict__ bias0, const float* __restrict__ bias1,
                 int ntStore) {
    __shared__ __align__(16) bf16_t As[128][40];    // [m][k], 80B row stride
    __shared__ __align__(16) bf16_t BsT[128][40];   // [n][k], 80B row stride

    const int tid  = threadIdx.x;
    const int lane = tid & 31, wid = tid >> 5;
    const int wm = wid >> 1, wn = wid & 1;          // 4x2 wave grid
    const int m0 = blockIdx.y * 128, n0 = blockIdx.x * 128;
    const int half = lane >> 4, l15 = lane & 15;

    v8f zero = {};
    v8f acc[2][4];
#pragma unroll
    for (int i = 0; i < 2; ++i)
#pragma unroll
        for (int j = 0; j < 4; ++j) acc[i][j] = zero;

    // staging coordinates
    const int ar = tid >> 1, ac = (tid & 1) * 16;   // A: 128 rows x 32 halves
    const int br = tid >> 3, bc = (tid & 7) * 16;   // B: 32 rows x 128 halves

    for (int kc = 0; kc < K; kc += 32) {
        // ---- A tile: 2 x 16B per thread, async global->LDS when available
        copy16B_to_lds(&As[ar][ac],     &A[(long)(m0 + ar) * K + kc + ac]);
        copy16B_to_lds(&As[ar][ac + 8], &A[(long)(m0 + ar) * K + kc + ac + 8]);

        // ---- B tile: vector global load, transposed scatter into LDS
        {
            const bf16_t* gb = &B[(long)(kc + br) * N + n0 + bc];
            union { bf16_t h[16]; uint4 q[2]; } u;
            u.q[0] = *(const uint4*)gb;
            u.q[1] = *(const uint4*)(gb + 8);
#pragma unroll
            for (int i = 0; i < 16; ++i) BsT[bc + i][br] = u.h[i];
        }
        if (kc + 32 < K)   // global_prefetch_b8 of next B tile
            __builtin_prefetch(&B[(long)(kc + 32 + br) * N + n0 + bc], 0, 1);

        async_copy_wait();
        __syncthreads();

        // ---- fragments (all 2x ds_load_b128 each)
        v16bf af[2], bfg[4];
#pragma unroll
        for (int i = 0; i < 2; ++i) {
            int row = wm * 32 + i * 16 + l15;
            union { v16bf v; uint4 q[2]; } u;
            u.q[0] = *(const uint4*)&As[row][half * 8];        // K = half*8+0..7
            u.q[1] = *(const uint4*)&As[row][16 + half * 8];   // K = 16+half*8+0..7
            af[i] = u.v;
        }
#pragma unroll
        for (int j = 0; j < 4; ++j) {
            int col = wn * 64 + j * 16 + l15;
            union { v16bf v; uint4 q[2]; } u;
            u.q[0] = *(const uint4*)&BsT[col][half * 8];       // K = half*8+0..7
            u.q[1] = *(const uint4*)&BsT[col][16 + half * 8];  // K = 16+half*8+0..7
            bfg[j] = u.v;
        }
#pragma unroll
        for (int i = 0; i < 2; ++i)
#pragma unroll
            for (int j = 0; j < 4; ++j)
                acc[i][j] = __builtin_amdgcn_wmma_f32_16x16x32_bf16(
                    false, af[i], false, bfg[j], (short)0, acc[i][j], false, false);
        __syncthreads();
    }

    // store D: row = base + half*8 + r, col = base + l15
#pragma unroll
    for (int i = 0; i < 2; ++i)
#pragma unroll
        for (int j = 0; j < 4; ++j) {
            int colg = n0 + wn * 64 + j * 16 + l15;
            float badd = 0.f;
            if (bias0) badd += bias0[colg];
            if (bias1) badd += bias1[colg];
#pragma unroll
            for (int r = 0; r < 8; ++r) {
                int rowg = m0 + wm * 32 + i * 16 + half * 8 + r;
                float v = acc[i][j][r] + badd;
                if (ntStore) __builtin_nontemporal_store(v, &C[(long)rowg * N + colg]);
                else C[(long)rowg * N + colg] = v;
            }
        }
}

// ---------------- clockwork-RNN scan: single WG, 32 waves ------------------
// wave w owns columns [w*32, w*32+32); h: fp32 in regs (D-frag layout),
// bf16 copy in LDS for A-fragments. WhT streamed from L2 (b128 per lane).
__global__ __launch_bounds__(1024)
void k_scan(const float* __restrict__ xWi, const bf16_t* __restrict__ WhT,
            const float* __restrict__ h0, bf16_t* __restrict__ h_all) {
    __shared__ __align__(16) bf16_t hB[16][NH + 8];

    const int tid  = threadIdx.x;
    const int lane = tid & 31, wid = tid >> 5;
    const int half = lane >> 4, l15 = lane & 15;
    const int ncol0 = wid * 32;

    float hreg[2][8];                                  // (col j, row r)
#pragma unroll
    for (int j = 0; j < 2; ++j) {
        int col = ncol0 + j * 16 + l15;
        float v = h0[col];
#pragma unroll
        for (int r = 0; r < 8; ++r) {
            hreg[j][r] = v;
            hB[half * 8 + r][col] = f2bf(v);
        }
    }
    __syncthreads();

    for (int t = 0; t < TSTEPS; ++t) {
        v8f zero = {};
        v8f acc[2]; acc[0] = zero; acc[1] = zero;

        for (int kc = 0; kc < NH; kc += 32) {
            v16bf af;
            {
                union { v16bf v; uint4 q[2]; } u;       // A row = batch = l15
                u.q[0] = *(const uint4*)&hB[l15][kc + half * 8];
                u.q[1] = *(const uint4*)&hB[l15][kc + 16 + half * 8];
                af = u.v;
            }
#pragma unroll
            for (int j = 0; j < 2; ++j) {
                int col = ncol0 + j * 16 + l15;
                union { v16bf v; uint4 q[2]; } u;       // contiguous K run in WhT
                const bf16_t* gp = &WhT[(long)col * NH + kc + half * 8];
                u.q[0] = *(const uint4*)gp;
                u.q[1] = *(const uint4*)(gp + 16);
                acc[j] = __builtin_amdgcn_wmma_f32_16x16x32_bf16(
                    false, af, false, u.v, (short)0, acc[j], false, false);
            }
        }
        __syncthreads();                               // everyone done reading hB

#pragma unroll
        for (int j = 0; j < 2; ++j) {
            int col = ncol0 + j * 16 + l15;
            int pm1 = (1 << (col >> 8)) - 1;           // clock period mask
            bool active = (t & pm1) == 0;
#pragma unroll
            for (int r = 0; r < 8; ++r) {
                int b = half * 8 + r;
                float pre = acc[j][r] + xWi[(long)(t * BATCH + b) * NH + col];
                float hn  = active ? tanhf(pre) : hreg[j][r];
                hreg[j][r] = hn;
                bf16_t hb = f2bf(hn);
                hB[b][col] = hb;
                h_all[((long)b * TSTEPS + t) * NH + col] = hb;  // (b,t) order
            }
        }
        __syncthreads();                               // hB visible for next step
    }
}

// ---------------- online log-sum-exp per logits row -----------------------
__global__ __launch_bounds__(256)
void k_lse(const float* __restrict__ logits, float* __restrict__ lse) {
    __shared__ float sm[256], ss[256];
    const int r = blockIdx.x, tid = threadIdx.x;
    const float* row = logits + (long)r * VOCAB;
    float m = -INFINITY, s = 0.f;
    for (int i = tid; i < VOCAB; i += 256) {
        float v = row[i];
        float nm = fmaxf(m, v);
        s = s * __expf(m - nm) + __expf(v - nm);
        m = nm;
    }
    sm[tid] = m; ss[tid] = s;
    __syncthreads();
    for (int off = 128; off > 0; off >>= 1) {
        if (tid < off) {
            float m2 = sm[tid + off], s2 = ss[tid + off];
            float nm = fmaxf(sm[tid], m2);
            ss[tid] = ss[tid] * __expf(sm[tid] - nm) + s2 * __expf(m2 - nm);
            sm[tid] = nm;
        }
        __syncthreads();
    }
    if (tid == 0) lse[r] = sm[0] + __logf(ss[0]);
}

// ---------------- per-row masked token log-prob ---------------------------
__global__ void k_rowll(const float* __restrict__ logits, const float* __restrict__ lse,
                        const int* __restrict__ x, const int* __restrict__ x_sl,
                        float* __restrict__ rowll) {
    int r = blockIdx.x * blockDim.x + threadIdx.x;
    if (r >= BATCH * TSTEPS) return;
    int b = r >> 8, t = r & 255;
    int tok = x[b * TFULL + t + 1];                    // y = x[:, 1:]
    float ll = logits[(long)r * VOCAB + tok] - lse[r];
    int sl = x_sl[b] - 1;
    rowll[r] = (t < sl) ? ll : 0.f;
}

// ---------------- deterministic final reduction ---------------------------
__global__ __launch_bounds__(256)
void k_loss(const float* __restrict__ rowll, const int* __restrict__ x_sl,
            float* __restrict__ out) {
    __shared__ float sb[256];
    int tid = threadIdx.x;
    float s = 0.f;
    for (int i = tid; i < BATCH * TSTEPS; i += 256) s += rowll[i];
    sb[tid] = s; __syncthreads();
    for (int off = 128; off > 0; off >>= 1) {
        if (tid < off) sb[tid] += sb[tid + off];
        __syncthreads();
    }
    if (tid == 0) {
        int sl = 0;
        for (int b = 0; b < BATCH; ++b) sl += x_sl[b] - 1;
        out[0] = -sb[0] / (float)sl;
    }
}

// ---------------- launcher -------------------------------------------------
extern "C" void kernel_launch(void* const* d_in, const int* in_sizes, int n_in,
                              void* d_out, int out_size, void* d_ws, size_t ws_size,
                              hipStream_t stream) {
    (void)in_sizes; (void)n_in; (void)out_size; (void)ws_size;
    const int*   x    = (const int*)  d_in[0];
    const int*   x_sl = (const int*)  d_in[1];
    const float* Wi   = (const float*)d_in[2];
    const float* Wh   = (const float*)d_in[3];
    const float* Wo   = (const float*)d_in[4];
    const float* bi   = (const float*)d_in[5];
    const float* bh   = (const float*)d_in[6];
    const float* h0   = (const float*)d_in[7];
    const float* emb  = (const float*)d_in[8];

    float* outp   = (float*)d_out;
    float* loss   = outp;                 // [0]
    float* logits = outp + 1;             // (B, T, VOCAB) fp32

    char* p = (char*)d_ws;
    auto carve = [&](size_t bytes) -> char* {
        char* q = p; p += (bytes + 255) & ~(size_t)255; return q;
    };
    bf16_t* Xbf  = (bf16_t*)carve((size_t)4096 * EMB * 2);
    bf16_t* WiB  = (bf16_t*)carve((size_t)EMB * NH * 2);
    bf16_t* WhT  = (bf16_t*)carve((size_t)NH * NH * 2);
    bf16_t* WoB  = (bf16_t*)carve((size_t)NH * VOCAB * 2);
    float*  xWi  = (float*) carve((size_t)4096 * NH * 4);
    bf16_t* hAll = (bf16_t*)carve((size_t)4096 * NH * 2);
    float*  lse  = (float*) carve((size_t)4096 * 4);
    float*  rll  = (float*) carve((size_t)4096 * 4);

    // 1. gather embeddings + bf16 weight conversion
    k_embed_gather<<<512, 256, 0, stream>>>(x, emb, Xbf);
    k_cvt8<<<256,  256, 0, stream>>>(Wi, WiB, (long)EMB * NH / 8);
    k_cvt_whT<<<1024, 256, 0, stream>>>(Wh, WhT);                 // mask + transpose
    k_cvt8<<<4096, 256, 0, stream>>>(Wo, WoB, (long)NH * VOCAB / 8);

    // 2. input projection hoisted out of scan: xWi = X @ Wi + bi + bh
    dim3 gC(NH / 128, 4096 / 128);
    k_gemm_bf16<<<gC, 256, 0, stream>>>(Xbf, WiB, xWi, 4096, NH, EMB, bi, bh, 0);

    // 3. serial clockwork scan (single WG, WMMA per step)
    k_scan<<<1, 1024, 0, stream>>>(xWi, WhT, h0, hAll);

    // 4. output GEMM: logits = h @ Wo  (write-bound; NT stores)
    dim3 gE(VOCAB / 128, 4096 / 128);
    k_gemm_bf16<<<gE, 256, 0, stream>>>(hAll, WoB, logits, 4096, VOCAB, NH,
                                        nullptr, nullptr, 1);

    // 5. log-softmax stats + masked NLL, deterministic reduction
    k_lse<<<4096, 256, 0, stream>>>(logits, lse);
    k_rowll<<<(4096 + 255) / 256, 256, 0, stream>>>(logits, lse, x, x_sl, rll);
    k_loss<<<1, 256, 0, stream>>>(rll, x_sl, loss);
}
